// ConsistencyUNet_60413009986055
// MI455X (gfx1250) — matrix operand order repaired and checked by reading, not compiled, not run on hardware
//
#include <hip/hip_runtime.h>

// ---------------------------------------------------------------------------
// Fused ConsistencyUNet forward for gfx1250 (MI455X).
//   kernel 1 (prep): fp32 params -> f16 GEMM weight tables + packed fp32 bias/GN
//   kernel 2 (main): per-wave fully fused UNet, all matmuls via
//                    v_wmma_f32_16x16x32_f16, activations staged in LDS.
// Round 2: full inlining (keeps address spaces -> global_load/ds_load instead
// of flat_load + full-drain waits), unified block schedule loop, B-fragment
// double buffering, waves_per_eu hint.
// ---------------------------------------------------------------------------

typedef _Float16 half8  __attribute__((ext_vector_type(8)));
typedef _Float16 half16 __attribute__((ext_vector_type(16)));
typedef float    floatx8 __attribute__((ext_vector_type(8)));

union H16 { half16 v; half8 h[2]; };

// ---- model dims -----------------------------------------------------------
#define HCH 128
#define SED 64
#define OBSD 60
#define ACTD 6

// ---- f16 weight table offsets (in halfs) ---------------------------------
#define OH_WIN   0          // 128 x 96  (inw, K padded 66->96)
#define OH_WS1   12288      // 128 x 64
#define OH_WS2   20480      // 64  x 128
#define OH_BLK   28672      // 7 blocks x [W1 128x128][W2 128x128][Wsp 128x64]
#define BLK_H    40960
#define OH_UP    315392     // 3 x 128x256
#define OH_WOUT  413696     // 16 x 128 (outw padded 6->16 rows)
#define NH_TOTAL 415744
#define FOFF_BYTES 831488   // = NH_TOTAL*2, 16B aligned

// ---- fp32 param table offsets (in floats) ---------------------------------
#define OF_S1B  0
#define OF_S2B  128
#define OF_INB  192
#define OF_BLK  320          // per block: n1w n1b c1b n2w n2b c2b spb (7*128)
#define BLK_F   896
#define OF_UPB  6592         // 3 x 128
#define OF_ONW  6976
#define OF_ONB  7104
#define OF_OUTB 7232         // padded to 16

// ---- per-wave LDS layout (bytes) ------------------------------------------
#define XSTR  132            // fp32 row stride (floats)
#define ASTR  264            // f16 A-stage row stride (halfs)
#define SESTR 72             // f16 se row stride (halfs)
#define SKSTR 136            // f16 skip row stride (halfs)
#define OFF_X   0
#define OFF_H   8448
#define OFF_A   16896
#define OFF_SE  25344
#define OFF_SK  27648
#define SKIP_BYTES 4352      // 16*136*2
#define WAVE_BYTES 40704
#define SMEM_BYTES (4*WAVE_BYTES)   // 4 waves / 128-thread block

__device__ __forceinline__ float siluf(float x) {
  return x * (1.0f / (1.0f + __expf(-x)));
}
__device__ __forceinline__ floatx8 vbcast(float v) {
  floatx8 r;
#pragma unroll
  for (int i = 0; i < 8; ++i) r[i] = v;
  return r;
}

// ---------------------------------------------------------------------------
// WMMA GEMM: D += A(16xK from LDS f16) * B(Kx(NT*16) from global f16 W, row =
// output channel, row-major with row stride wstr halfs, starting column wk0).
// A layout used here (contiguous k, same for B): lanes 0-15 hold K0-15,
// lanes 16-31 hold K16-31. B: lane n = col n, K(sel*16 .. +15).
// B fragments are double-buffered so a global load is in flight past each WMMA.
// ---------------------------------------------------------------------------
template <int NT, int KS>
__device__ __forceinline__ void wmma_gemm(floatx8* acc, const _Float16* a, int astr,
                                          const _Float16* __restrict__ W, int wstr,
                                          int wk0, int lane) {
  const int sel  = lane >> 4;
  const int l15  = lane & 15;
  H16 A[KS];
#pragma unroll
  for (int ks = 0; ks < KS; ++ks) {
    const _Float16* ap = a + l15 * astr + ks * 32 + sel * 16;
    A[ks].h[0] = *(const half8*)(ap);
    A[ks].h[1] = *(const half8*)(ap + 8);
  }
  const _Float16* __restrict__ wbase = W + (size_t)l15 * wstr + wk0 + sel * 16;
  H16 Bcur, Bnxt;
  Bcur.h[0] = *(const half8*)(wbase);
  Bcur.h[1] = *(const half8*)(wbase + 8);
#pragma unroll
  for (int nt = 0; nt < NT; ++nt) {
#pragma unroll
    for (int ks = 0; ks < KS; ++ks) {
      int nks = (ks + 1 == KS) ? 0 : ks + 1;
      int nnt = (ks + 1 == KS) ? nt + 1 : nt;
      if (nnt < NT) {
        const _Float16* wp = wbase + (size_t)(nnt * 16) * wstr + nks * 32;
        Bnxt.h[0] = *(const half8*)(wp);
        Bnxt.h[1] = *(const half8*)(wp + 8);
      }
      acc[nt] = __builtin_amdgcn_wmma_f32_16x16x32_f16(
          false, A[ks].v, false, Bcur.v, (short)0, acc[nt], false, false);
      Bcur = Bnxt;
    }
  }
}

// GroupNorm(G=8 -> 16ch groups) + SiLU : fp32 LDS -> f16 LDS A-stage.
// 16 rows * 8 groups = 128 work items = 4 iters * 32 lanes.
__device__ __forceinline__ void gn_silu(const float* src, int sstr, _Float16* dst,
                                        int dstr, const float* __restrict__ gw,
                                        const float* __restrict__ gb, int lane) {
#pragma unroll
  for (int it = 0; it < 4; ++it) {
    int idx = it * 32 + lane;
    int row = idx >> 3, g = idx & 7;
    const float* p = src + row * sstr + g * 16;
    float x[16];
    float s = 0.f, s2 = 0.f;
#pragma unroll
    for (int c = 0; c < 16; ++c) { x[c] = p[c]; s += x[c]; s2 += x[c] * x[c]; }
    float mu  = s * 0.0625f;
    float var = s2 * 0.0625f - mu * mu;
    float rs  = rsqrtf(var + 1e-5f);
#pragma unroll
    for (int c = 0; c < 16; ++c) {
      float xn = (x[c] - mu) * rs * gw[g * 16 + c] + gb[g * 16 + c];
      dst[row * dstr + g * 16 + c] = (_Float16)siluf(xn);
    }
  }
}

// store C/D-layout accumulators to fp32 LDS (lane&15 = col, rows sel*8..+7)
template <int NT>
__device__ __forceinline__ void acc_store_f32(float* buf, int str,
                                              const floatx8* acc, int lane) {
  const int c0 = lane & 15, rb = (lane >> 4) * 8;
#pragma unroll
  for (int nt = 0; nt < NT; ++nt)
#pragma unroll
    for (int i = 0; i < 8; ++i)
      buf[(rb + i) * str + nt * 16 + c0] = acc[nt][i];
}

// one residual block on this wave's 16 rows (forceinline: keep addr spaces)
__device__ __forceinline__ void resblock(float* xbuf, float* hbuf,
                                         _Float16* astage,
                                         const _Float16* sebuf,
                                         const _Float16* __restrict__ Wblk,
                                         const float* __restrict__ Fblk, int lane) {
  const int c0 = lane & 15, rb = (lane >> 4) * 8;
  // t = silu(gn(x))
  gn_silu(xbuf, XSTR, astage, ASTR, Fblk + 0, Fblk + 128, lane);
  __syncthreads();
  // h = t @ W1.T + c1b + se @ Wsp.T + spb
  floatx8 acc[8];
#pragma unroll
  for (int nt = 0; nt < 8; ++nt)
    acc[nt] = vbcast(Fblk[256 + nt * 16 + c0] + Fblk[768 + nt * 16 + c0]);
  wmma_gemm<8, 4>(acc, astage, ASTR, Wblk, 128, 0, lane);
  wmma_gemm<8, 2>(acc, sebuf, SESTR, Wblk + 2 * 16384, 64, 0, lane);
  acc_store_f32<8>(hbuf, XSTR, acc, lane);
  __syncthreads();
  // u = silu(gn(h))
  gn_silu(hbuf, XSTR, astage, ASTR, Fblk + 384, Fblk + 512, lane);
  __syncthreads();
  // x += u @ W2.T + c2b
  floatx8 acc2[8];
#pragma unroll
  for (int nt = 0; nt < 8; ++nt) acc2[nt] = vbcast(Fblk[640 + nt * 16 + c0]);
  wmma_gemm<8, 4>(acc2, astage, ASTR, Wblk + 16384, 128, 0, lane);
#pragma unroll
  for (int nt = 0; nt < 8; ++nt)
#pragma unroll
    for (int i = 0; i < 8; ++i)
      xbuf[(rb + i) * XSTR + nt * 16 + c0] += acc2[nt][i];
  __syncthreads();
}

// ---------------------------------------------------------------------------
__global__ __launch_bounds__(128)
__attribute__((amdgpu_waves_per_eu(2)))
void unet_main(
    const float* __restrict__ obs, const float* __restrict__ act,
    const float* __restrict__ sigma, const _Float16* __restrict__ wsh,
    const float* __restrict__ wsf, float* __restrict__ out, int Bn) {
  extern __shared__ char smem[];
  const int lane = threadIdx.x & 31;
  const int wave = threadIdx.x >> 5;
  const int rowbase = blockIdx.x * 64 + wave * 16;
  char* wb = smem + wave * WAVE_BYTES;
  float*     xbuf   = (float*)(wb + OFF_X);
  float*     hbuf   = (float*)(wb + OFF_H);
  _Float16*  astage = (_Float16*)(wb + OFF_A);
  _Float16*  sebuf  = (_Float16*)(wb + OFF_SE);
  _Float16*  skips  = (_Float16*)(wb + OFF_SK);
  const int c0 = lane & 15, rb = (lane >> 4) * 8;

  // ---- sigma embedding: emb[row][j] = sin/cos(sigma*freq) -> astage (16x64)
  const float LN = 9.210340371976184f;  // log(10000)
#pragma unroll 4
  for (int t = 0; t < 32; ++t) {
    int idx = t * 32 + lane;
    int row = idx >> 6, j = idx & 63, jj = j & 31;
    int g = rowbase + row; if (g >= Bn) g = Bn - 1;
    float f = __expf(-LN * (float)jj * (1.0f / 31.0f));
    float a = sigma[g] * f;
    astage[row * ASTR + j] = (_Float16)((j < 32) ? __sinf(a) : __cosf(a));
  }
  __syncthreads();
  // ---- se = (silu(emb @ s1w.T + s1b)) @ s2w.T + s2b
  floatx8 acc[8];
#pragma unroll
  for (int nt = 0; nt < 8; ++nt) acc[nt] = vbcast(wsf[OF_S1B + nt * 16 + c0]);
  wmma_gemm<8, 2>(acc, astage, ASTR, wsh + OH_WS1, 64, 0, lane);
  __syncthreads();
#pragma unroll
  for (int nt = 0; nt < 8; ++nt)
#pragma unroll
    for (int i = 0; i < 8; ++i)
      astage[(rb + i) * ASTR + nt * 16 + c0] = (_Float16)siluf(acc[nt][i]);
  __syncthreads();
  floatx8 se4[4];
#pragma unroll
  for (int nt = 0; nt < 4; ++nt) se4[nt] = vbcast(wsf[OF_S2B + nt * 16 + c0]);
  wmma_gemm<4, 4>(se4, astage, ASTR, wsh + OH_WS2, 128, 0, lane);
  __syncthreads();
#pragma unroll
  for (int nt = 0; nt < 4; ++nt)
#pragma unroll
    for (int i = 0; i < 8; ++i)
      sebuf[(rb + i) * SESTR + nt * 16 + c0] = (_Float16)se4[nt][i];

  // ---- input projection: concat(obs,act) (16x66 pad 96) @ inw.T + inb
  for (int t = 0; t < 48; ++t) {
    int idx = t * 32 + lane;
    int row = idx / 96, k = idx % 96;
    int g = rowbase + row; if (g >= Bn) g = Bn - 1;
    float v = (k < OBSD) ? obs[g * OBSD + k]
            : (k < OBSD + ACTD) ? act[g * ACTD + (k - OBSD)] : 0.0f;
    astage[row * ASTR + k] = (_Float16)v;
  }
  __syncthreads();
#pragma unroll
  for (int nt = 0; nt < 8; ++nt) acc[nt] = vbcast(wsf[OF_INB + nt * 16 + c0]);
  wmma_gemm<8, 3>(acc, astage, ASTR, wsh + OH_WIN, 96, 0, lane);
  acc_store_f32<8>(xbuf, XSTR, acc, lane);
  __syncthreads();

  // ---- unified schedule: step 0..2 = down1..3, 3 = mid, 4..6 = up3..1
#pragma unroll 1
  for (int step = 0; step < 7; ++step) {
    if (step >= 4) {
      // up_proj: x = concat(x, skip[6-step]) @ Wup.T + upb
      int j = step - 4;
#pragma unroll 4
      for (int t = 0; t < 64; ++t) {   // stage x -> f16
        int idx = t * 32 + lane;
        int row = idx >> 7, c = idx & 127;
        astage[row * ASTR + c] = (_Float16)xbuf[row * XSTR + c];
      }
      __syncthreads();
      floatx8 uacc[8];
#pragma unroll
      for (int nt = 0; nt < 8; ++nt)
        uacc[nt] = vbcast(wsf[OF_UPB + j * 128 + nt * 16 + c0]);
      const _Float16* __restrict__ Wup = wsh + OH_UP + j * 32768;
      wmma_gemm<8, 4>(uacc, astage, ASTR, Wup, 256, 0, lane);
      wmma_gemm<8, 4>(uacc, skips + (2 - j) * (SKIP_BYTES / 2), SKSTR, Wup, 256,
                      128, lane);
      __syncthreads();
      acc_store_f32<8>(xbuf, XSTR, uacc, lane);
      __syncthreads();
    }
    resblock(xbuf, hbuf, astage, sebuf, wsh + OH_BLK + step * BLK_H,
             wsf + OF_BLK + step * BLK_F, lane);
    if (step < 3) {
      _Float16* sk = skips + step * (SKIP_BYTES / 2);
#pragma unroll 4
      for (int t = 0; t < 64; ++t) {
        int idx = t * 32 + lane;
        int row = idx >> 7, c = idx & 127;
        sk[row * SKSTR + c] = (_Float16)xbuf[row * XSTR + c];
      }
      __syncthreads();
    }
  }

  // ---- head: silu(gn(x)) @ outw.T + outb
  gn_silu(xbuf, XSTR, astage, ASTR, wsf + OF_ONW, wsf + OF_ONB, lane);
  __syncthreads();
  floatx8 oacc = vbcast(wsf[OF_OUTB + c0]);
  wmma_gemm<1, 4>(&oacc, astage, ASTR, wsh + OH_WOUT, 128, 0, lane);
  if (c0 < ACTD) {
#pragma unroll
    for (int i = 0; i < 8; ++i) {
      int gr = rowbase + rb + i;
      if (gr < Bn) out[gr * ACTD + c0] = oacc[i];
    }
  }
}

// ---------------------------------------------------------------------------
// Prep: convert/pack all params into workspace. 86 jobs decoded from job id.
// d_in flatten assumption: depth-first insertion order of setup_inputs().
// ---------------------------------------------------------------------------
struct PtrTab { const float* p[89]; };

__device__ __forceinline__ void decode_job(int jid, int& src, int& dst, int& Rd,
                                           int& Cd, int& Rs, int& Cs, int& es,
                                           int& eo, int& isH) {
  const int bsrc[7] = {9, 19, 29, 39, 51, 63, 75};
  es = 1; eo = 0; isH = 1; Rd = 1; Rs = 1;
  if (jid == 0)      { src = 7; dst = OH_WIN; Rd = 128; Cd = 96;  Rs = 128; Cs = 66; }
  else if (jid == 1) { src = 3; dst = OH_WS1; Rd = 128; Cd = 64;  Rs = 128; Cs = 64; }
  else if (jid == 2) { src = 5; dst = OH_WS2; Rd = 64;  Cd = 128; Rs = 64;  Cs = 128; }
  else if (jid < 24) {
    int b = (jid - 3) / 3, t = (jid - 3) % 3, sb = bsrc[b];
    int base = OH_BLK + b * BLK_H;
    if (t == 0)      { src = sb + 2; dst = base;         Rd = 128; Cd = 128; Rs = 128; Cs = 128; es = 3; eo = 1; }
    else if (t == 1) { src = sb + 6; dst = base + 16384; Rd = 128; Cd = 128; Rs = 128; Cs = 128; es = 3; eo = 1; }
    else             { src = sb + 8; dst = base + 32768; Rd = 128; Cd = 64;  Rs = 128; Cs = 64; }
  } else if (jid < 27) {
    const int us[3] = {49, 61, 73};
    int j = jid - 24;
    src = us[j]; dst = OH_UP + j * 32768; Rd = 128; Cd = 256; Rs = 128; Cs = 256;
  } else if (jid == 27) { src = 87; dst = OH_WOUT; Rd = 16; Cd = 128; Rs = 6; Cs = 128; }
  else {
    isH = 0;
    int f = jid - 28;
    if (f == 0)      { src = 4; dst = OF_S1B; Cd = 128; Cs = 128; }
    else if (f == 1) { src = 6; dst = OF_S2B; Cd = 64;  Cs = 64; }
    else if (f == 2) { src = 8; dst = OF_INB; Cd = 128; Cs = 128; }
    else if (f < 52) {
      int k = f - 3, b = k / 7, t = k % 7;
      const int toff[7] = {0, 1, 3, 4, 5, 7, 9};
      src = bsrc[b] + toff[t]; dst = OF_BLK + b * BLK_F + t * 128; Cd = 128; Cs = 128;
    } else if (f < 55) {
      const int us[3] = {50, 62, 74};
      int j = f - 52;
      src = us[j]; dst = OF_UPB + j * 128; Cd = 128; Cs = 128;
    }
    else if (f == 55) { src = 85; dst = OF_ONW;  Cd = 128; Cs = 128; }
    else if (f == 56) { src = 86; dst = OF_ONB;  Cd = 128; Cs = 128; }
    else              { src = 88; dst = OF_OUTB; Cd = 16;  Cs = 6; }
  }
}

__global__ void unet_prep(PtrTab tab, _Float16* __restrict__ wsh,
                          float* __restrict__ wsf) {
  int src, dst, Rd, Cd, Rs, Cs, es, eo, isH;
  decode_job(blockIdx.x, src, dst, Rd, Cd, Rs, Cs, es, eo, isH);
  int n = Rd * Cd;
  int e = blockIdx.y * 256 + threadIdx.x;
  if (e >= n) return;
  int r = e / Cd, c = e % Cd;
  float v = 0.0f;
  if (r < Rs && c < Cs) v = tab.p[src][(r * Cs + c) * es + eo];
  if (isH) wsh[dst + e] = (_Float16)v;
  else     wsf[dst + e] = v;
}

// ---------------------------------------------------------------------------
extern "C" void kernel_launch(void* const* d_in, const int* in_sizes, int n_in,
                              void* d_out, int out_size, void* d_ws, size_t ws_size,
                              hipStream_t stream) {
  PtrTab tab{};
  for (int i = 0; i < 89 && i < n_in; ++i) tab.p[i] = (const float*)d_in[i];

  _Float16* wsh = (_Float16*)d_ws;
  float*    wsf = (float*)((char*)d_ws + FOFF_BYTES);
  const float* obs = (const float*)d_in[0];
  const float* act = (const float*)d_in[1];
  const float* sig = (const float*)d_in[2];
  const int Bn = in_sizes[2];

  unet_prep<<<dim3(86, 128, 1), 256, 0, stream>>>(tab, wsh, wsf);

  (void)hipFuncSetAttribute((const void*)unet_main,
                            hipFuncAttributeMaxDynamicSharedMemorySize,
                            SMEM_BYTES);
  dim3 grid((Bn + 63) / 64);
  unet_main<<<grid, 128, SMEM_BYTES, stream>>>(obs, act, sig, wsh, wsf,
                                               (float*)d_out, Bn);
}
